// Graph_conv_11381663334637
// MI455X (gfx1250) — matrix operand order repaired and checked by reading, not compiled, not run on hardware
//
#include <hip/hip_runtime.h>
#include <cstdint>
#include <cstddef>

// ---------------------------------------------------------------------------
// Types for CDNA5 WMMA (wave32): v_wmma_f32_16x16x32_bf16
// ---------------------------------------------------------------------------
typedef __attribute__((ext_vector_type(16))) __bf16 v16bf;
typedef __attribute__((ext_vector_type(8)))  float  v8f;

#define EPSF   1e-12f
#define BN_EPS 1e-5f

// ---- CDNA5 async memory->LDS helpers (ISA 15.18.3 op 98, 15.5 op 74) ------
__device__ __forceinline__ uint32_t lds_off_u32(const void* p) {
  // flat->LDS mapping truncates to addr[31:0] (ISA 10.2), so the low 32 bits
  // of the generic pointer are the LDS byte address.
  return (uint32_t)(uintptr_t)p;
}
__device__ __forceinline__ void async_load_b128_to_lds(uint32_t lds, uint64_t gaddr) {
  asm volatile("global_load_async_to_lds_b128 %0, %1, off"
               :: "v"(lds), "v"(gaddr)
               : "memory");
}
__device__ __forceinline__ void wait_asynccnt0() {
  asm volatile("s_wait_asynccnt 0" ::: "memory");
}

// ======================= elementwise / conversion ==========================
__global__ void k_convert(const float* __restrict__ in, __bf16* __restrict__ out, int n) {
  int i = blockIdx.x * 256 + threadIdx.x;
  if (i < n) out[i] = (__bf16)in[i];
}

__global__ void k_convert_strided(const float* __restrict__ in, int ldin,
                                  __bf16* __restrict__ out, int ldout,
                                  int rows, int cols) {
  int i = blockIdx.x * 256 + threadIdx.x;
  if (i < rows * cols) {
    int r = i / cols, c = i - r * cols;
    out[(size_t)r * ldout + c] = (__bf16)in[(size_t)r * ldin + c];
  }
}

__global__ void k_transpose_to_bf16(const float* __restrict__ in, __bf16* __restrict__ out,
                                    int rows, int cols) {
  int i = blockIdx.x * 256 + threadIdx.x;
  if (i < rows * cols) {
    int r = i / cols, c = i - r * cols;
    out[(size_t)c * rows + r] = (__bf16)in[i];
  }
}

// ======================= BatchNorm2d (train stats) =========================
// y1: (2,256,4096) flat. One block per channel reduces 8192 samples.
__global__ void k_bn2d_reduce(const float* __restrict__ y, float* __restrict__ mv) {
  __shared__ float ss[256], sq[256];
  int c = blockIdx.x;
  float s = 0.f, q = 0.f;
  for (int i = threadIdx.x; i < 8192; i += 256) {
    int b = i >> 12, p = i & 4095;
    float v = y[(size_t)b * 1048576 + (size_t)c * 4096 + p];
    s += v; q += v * v;
  }
  ss[threadIdx.x] = s; sq[threadIdx.x] = q;
  __syncthreads();
  for (int st = 128; st > 0; st >>= 1) {
    if (threadIdx.x < st) { ss[threadIdx.x] += ss[threadIdx.x + st]; sq[threadIdx.x] += sq[threadIdx.x + st]; }
    __syncthreads();
  }
  if (threadIdx.x == 0) {
    float m = ss[0] * (1.0f / 8192.0f);
    mv[c] = m;
    mv[256 + c] = sq[0] * (1.0f / 8192.0f) - m * m;
  }
}

__global__ void k_bn2d_apply(float* __restrict__ y, const float* __restrict__ mv,
                             const float* __restrict__ g, const float* __restrict__ b) {
  int i = blockIdx.x * 256 + threadIdx.x;
  if (i < 2097152) {
    int c = (i >> 12) & 255;
    float r = (y[i] - mv[c]) * rsqrtf(mv[256 + c] + BN_EPS) * g[c] + b[c];
    y[i] = fmaxf(r, 0.f);
  }
}

// ================= fused upsample(512)->mask->downsample(64) ===============
// out[c,i,j] = sum over 4 corners (512-grid) of w * mask * bilinear64(feat)
__global__ void k_posnode(const float* __restrict__ feat,   // (256,64,64)
                          const float* __restrict__ mask,   // (512,512)
                          float* __restrict__ pos) {        // (256,64,64)
  int i = blockIdx.x * 256 + threadIdx.x;
  if (i >= 1048576) return;
  int c = i >> 12;
  int p = i & 4095;
  int iy = p >> 6, ix = p & 63;
  const float* F = feat + (size_t)c * 4096;
  const float SD = 511.0f / 63.0f;
  const float SU = 63.0f / 511.0f;
  float sy = iy * SD, sx = ix * SD;
  int Y0 = (int)floorf(sy); if (Y0 > 511) Y0 = 511;
  int Y1 = Y0 + 1;          if (Y1 > 511) Y1 = 511;
  int X0 = (int)floorf(sx); if (X0 > 511) X0 = 511;
  int X1 = X0 + 1;          if (X1 > 511) X1 = 511;
  float wy = sy - (float)Y0, wx = sx - (float)X0;
  int   Ys[2] = {Y0, Y1};           int   Xs[2] = {X0, X1};
  float Wy[2] = {1.f - wy, wy};     float Wx[2] = {1.f - wx, wx};
  float acc = 0.f;
#pragma unroll
  for (int a = 0; a < 2; ++a) {
    int Y = Ys[a];
    float uy = Y * SU;
    int y0 = (int)floorf(uy); if (y0 > 63) y0 = 63;
    int y1 = y0 + 1;          if (y1 > 63) y1 = 63;
    float fy = uy - (float)y0;
#pragma unroll
    for (int bq = 0; bq < 2; ++bq) {
      int X = Xs[bq];
      float uxp = X * SU;
      int x0 = (int)floorf(uxp); if (x0 > 63) x0 = 63;
      int x1 = x0 + 1;           if (x1 > 63) x1 = 63;
      float fx = uxp - (float)x0;
      float q = F[y0 * 64 + x0] * (1.f - fy) * (1.f - fx)
              + F[y0 * 64 + x1] * (1.f - fy) * fx
              + F[y1 * 64 + x0] * fy * (1.f - fx)
              + F[y1 * 64 + x1] * fy * fx;
      acc += Wy[a] * Wx[bq] * mask[Y * 512 + X] * q;
    }
  }
  pos[i] = acc;
}

// =================== per-column reductions (node axis) =====================
__global__ void k_colsumsq(const float* __restrict__ in, float* __restrict__ out,
                           int cols, int rpc) {
  int c = blockIdx.x * 256 + threadIdx.x;
  if (c >= cols) return;
  size_t r0 = (size_t)blockIdx.y * rpc;
  float s = 0.f;
  for (size_t r = r0; r < r0 + (size_t)rpc; ++r) { float v = in[r * cols + c]; s += v * v; }
  atomicAdd(&out[c], s);
}

__global__ void k_colabssum(const float* __restrict__ in, float* __restrict__ out,
                            int cols, int rpc) {
  int c = blockIdx.x * 256 + threadIdx.x;
  if (c >= cols) return;
  size_t r0 = (size_t)blockIdx.y * rpc;
  float s = 0.f;
  for (size_t r = r0; r < r0 + (size_t)rpc; ++r) s += fabsf(in[r * cols + c]);
  atomicAdd(&out[c], s);
}

__global__ void k_colnorm_apply(const float* __restrict__ in, const float* __restrict__ ss,
                                __bf16* __restrict__ out, int n, int colmask) {
  int i = blockIdx.x * 256 + threadIdx.x;
  if (i < n) {
    int c = i & colmask;
    float nv = fmaxf(sqrtf(ss[c]), EPSF);
    out[i] = (__bf16)(in[i] / nv);
  }
}

// An = A / clip(colabssum, eps), stored bf16 (A.max() cancels here)
__global__ void k_an_convert(const float* __restrict__ A, const float* __restrict__ cs,
                             __bf16* __restrict__ out, int n) {
  int i = blockIdx.x * 256 + threadIdx.x;
  if (i < n) {
    int c = i & 4095;
    out[i] = (__bf16)(A[i] / fmaxf(cs[c], EPSF));
  }
}

// =========================== max reduction for R ===========================
__global__ void k_max_partial(const float* __restrict__ in, float* __restrict__ part, int n) {
  __shared__ float sm[256];
  float m = -3.0e38f;
  for (int i = blockIdx.x * 256 + threadIdx.x; i < n; i += gridDim.x * 256) m = fmaxf(m, in[i]);
  sm[threadIdx.x] = m; __syncthreads();
  for (int st = 128; st > 0; st >>= 1) {
    if (threadIdx.x < st) sm[threadIdx.x] = fmaxf(sm[threadIdx.x], sm[threadIdx.x + st]);
    __syncthreads();
  }
  if (threadIdx.x == 0) part[blockIdx.x] = sm[0];
}

__global__ void k_max_final(const float* __restrict__ part, float* __restrict__ out, int n) {
  __shared__ float sm[256];
  float m = -3.0e38f;
  for (int i = threadIdx.x; i < n; i += 256) m = fmaxf(m, part[i]);
  sm[threadIdx.x] = m; __syncthreads();
  for (int st = 128; st > 0; st >>= 1) {
    if (threadIdx.x < st) sm[threadIdx.x] = fmaxf(sm[threadIdx.x], sm[threadIdx.x + st]);
    __syncthreads();
  }
  if (threadIdx.x == 0) out[0] = sm[0];
}

// RT_bf16[m,n] = (R[n,m]/R.max)   (only the transpose is consumed)
__global__ void k_rT_convert(const float* __restrict__ R, const float* __restrict__ rmax,
                             __bf16* __restrict__ RTb) {
  int i = blockIdx.x * 256 + threadIdx.x;
  if (i < 16777216) {
    float inv = 1.0f / rmax[0];
    int n = i >> 12, m = i & 4095;
    RTb[(size_t)m * 4096 + n] = (__bf16)(R[i] * inv);
  }
}

// ============================ BatchNorm1d ==================================
__global__ void k_bn1d_reduce(const float* __restrict__ y, float* __restrict__ s1,
                              float* __restrict__ s2) {
  int c = blockIdx.x * 256 + threadIdx.x;   // cols = 512, grid.x = 2
  int r0 = blockIdx.y * 256;                // grid.y = 16
  float s = 0.f, q = 0.f;
  for (int r = r0; r < r0 + 256; ++r) { float v = y[(size_t)r * 512 + c]; s += v; q += v * v; }
  atomicAdd(&s1[c], s); atomicAdd(&s2[c], q);
}

__global__ void k_bn1d_apply(float* __restrict__ y, const float* __restrict__ s1,
                             const float* __restrict__ s2, const float* __restrict__ g,
                             const float* __restrict__ b) {
  int i = blockIdx.x * 256 + threadIdx.x;
  if (i < 2097152) {
    int c = i & 511;
    float m = s1[c] * (1.0f / 4096.0f);
    float v = s2[c] * (1.0f / 4096.0f) - m * m;
    float r = (y[i] - m) * rsqrtf(v + BN_EPS) * g[c] + b[c];
    y[i] = fmaxf(r, 0.f);
  }
}

// ======================= generic bf16 WMMA GEMM ============================
// C(MxN) = A(MxK) * op(B) [+bias[col]] [+addm] [relu], store f32 and/or bf16(T)
// bT=1: B is (NxK) row-major (C = A*B^T).  bT=0: B is (KxN) row-major.
// M,N multiples of 128; K multiple of 32; lda/ldb multiples of 8.
#define EPI_BIAS 1
#define EPI_RELU 2
#define EPI_ADD  4
#define EPI_F32  8
#define EPI_B16  16
#define EPI_B16T 32

#define GTM 128
#define GTN 128
#define GTK 32
#define GLS 40   // LDS row stride (bf16) with padding

__global__ __launch_bounds__(256)
void k_gemm_bf16(const __bf16* __restrict__ A, int lda,
                 const __bf16* __restrict__ B, int ldb, int bT,
                 float* __restrict__ Cf, int ldc,
                 __bf16* __restrict__ Cb, int ldcb,
                 const float* __restrict__ bias,
                 const float* __restrict__ addm, int ldadd,
                 int K, int flags)
{
  __shared__ __align__(16) __bf16 sA[GTM * GLS];
  __shared__ __align__(16) __bf16 sB[GTN * GLS];

  const int tid  = threadIdx.x;
  const int lane = tid & 31;
  const int wave = tid >> 5;     // 8 waves
  const int wm   = wave >> 1;    // 0..3 : 32-row strip
  const int wn   = wave & 1;     // 0..1 : 64-col strip
  const int l15  = lane & 15;
  const int kh   = lane >> 4;

  const size_t tileM = (size_t)blockIdx.y * GTM;
  const size_t tileN = (size_t)blockIdx.x * GTN;

  v8f acc[2][4];
#pragma unroll
  for (int mi = 0; mi < 2; ++mi)
#pragma unroll
    for (int ni = 0; ni < 4; ++ni)
      acc[mi][ni] = (v8f){0.f, 0.f, 0.f, 0.f, 0.f, 0.f, 0.f, 0.f};

  for (int k0 = 0; k0 < K; k0 += GTK) {
    // ---- stage A tile 128x32 via async memory->LDS (ASYNCcnt path) ----
#pragma unroll
    for (int it = 0; it < 2; ++it) {
      int i  = tid + it * 256;
      int r  = i >> 2;
      int kc = (i & 3) << 3;
      async_load_b128_to_lds(lds_off_u32(&sA[r * GLS + kc]),
          (uint64_t)(uintptr_t)&A[(tileM + (size_t)r) * (size_t)lda + k0 + kc]);
    }
    // prefetch next A K-panel into L2 (global_prefetch_b8)
    if (k0 + GTK < K)
      __builtin_prefetch(&A[(tileM + (size_t)(tid >> 1)) * (size_t)lda + k0 + GTK], 0, 1);

    // ---- stage B tile into canonical [n][k] layout ----
    if (bT) {
#pragma unroll
      for (int it = 0; it < 2; ++it) {
        int i  = tid + it * 256;
        int r  = i >> 2;
        int kc = (i & 3) << 3;
        async_load_b128_to_lds(lds_off_u32(&sB[r * GLS + kc]),
            (uint64_t)(uintptr_t)&B[(tileN + (size_t)r) * (size_t)ldb + k0 + kc]);
      }
    } else {
      // (KxN) -> [n][k] transpose: needs element scatter, keep sync path
#pragma unroll
      for (int it = 0; it < 2; ++it) {
        int i  = tid + it * 256;
        int kr = i >> 4;            // 0..31
        int nc = (i & 15) << 3;     // 0..120
        uint4 v = *(const uint4*)(&B[(size_t)(k0 + kr) * (size_t)ldb + tileN + nc]);
        const __bf16* h = (const __bf16*)&v;
#pragma unroll
        for (int e = 0; e < 8; ++e) sB[(nc + e) * GLS + kr] = h[e];
      }
    }
    wait_asynccnt0();     // each wave drains its own async loads ...
    __syncthreads();      // ... then the barrier publishes LDS to all waves

    // ---- fragments per ISA 7.12.2 layouts ----
    v16bf af[2], bf[4];
#pragma unroll
    for (int mi = 0; mi < 2; ++mi) {
      const __bf16* pa = &sA[(wm * 32 + mi * 16 + l15) * GLS];
#pragma unroll
      for (int e = 0; e < 8; ++e) {
        af[mi][e]     = pa[kh * 8 + e];        // K = kh*8 + e
        af[mi][8 + e] = pa[16 + kh * 8 + e];   // K = 16 + kh*8 + e
      }
    }
#pragma unroll
    for (int ni = 0; ni < 4; ++ni) {
      const __bf16* pb = &sB[(wn * 64 + ni * 16 + l15) * GLS];
#pragma unroll
      for (int e = 0; e < 8; ++e) {
        bf[ni][e]     = pb[kh * 8 + e];        // K = kh*8 + e
        bf[ni][8 + e] = pb[16 + kh * 8 + e];   // K = 16 + kh*8 + e
      }
    }

#pragma unroll
    for (int mi = 0; mi < 2; ++mi)
#pragma unroll
      for (int ni = 0; ni < 4; ++ni)
        acc[mi][ni] = __builtin_amdgcn_wmma_f32_16x16x32_bf16(
            false, af[mi], false, bf[ni], (short)0, acc[mi][ni], false, false);

    __syncthreads();
  }

  // ---- epilogue: C/D layout row = v + 8*kh, col = l15 ----
#pragma unroll
  for (int mi = 0; mi < 2; ++mi) {
#pragma unroll
    for (int ni = 0; ni < 4; ++ni) {
      size_t col = tileN + wn * 64 + ni * 16 + l15;
      float bv = (flags & EPI_BIAS) ? bias[col] : 0.f;
#pragma unroll
      for (int v = 0; v < 8; ++v) {
        size_t row = tileM + wm * 32 + mi * 16 + kh * 8 + v;
        float val = acc[mi][ni][v] + bv;
        if (flags & EPI_ADD)  val += addm[row * (size_t)ldadd + col];
        if (flags & EPI_RELU) val = fmaxf(val, 0.f);
        if (flags & EPI_F32)  Cf[row * (size_t)ldc + col] = val;
        if (flags & EPI_B16) {
          if (flags & EPI_B16T) Cb[col * (size_t)ldcb + row] = (__bf16)val;
          else                  Cb[row * (size_t)ldcb + col] = (__bf16)val;
        }
      }
    }
  }
}

// ===========================================================================
extern "C" void kernel_launch(void* const* d_in, const int* in_sizes, int n_in,
                              void* d_out, int out_size, void* d_ws, size_t ws_size,
                              hipStream_t stream) {
  (void)in_sizes; (void)n_in; (void)out_size; (void)ws_size;
  const float* x      = (const float*)d_in[0];
  const float* mask   = (const float*)d_in[1];
  const float* conv1w = (const float*)d_in[2];
  const float* bn1g   = (const float*)d_in[3];
  const float* bn1b   = (const float*)d_in[4];
  const float* g0aw   = (const float*)d_in[5];
  const float* g0ab   = (const float*)d_in[6];
  const float* g0uw   = (const float*)d_in[7];
  const float* g0ub   = (const float*)d_in[8];
  const float* g1aw   = (const float*)d_in[9];
  const float* g1ab   = (const float*)d_in[10];
  const float* g1uw   = (const float*)d_in[11];
  const float* g1ub   = (const float*)d_in[12];
  const float* cgw    = (const float*)d_in[13];
  const float* cgb    = (const float*)d_in[14];
  const float* cgbng  = (const float*)d_in[15];
  const float* cgbnb  = (const float*)d_in[16];

  char* base = (char*)d_ws;
  size_t off = 0;
  auto alloc = [&](size_t bytes) -> void* {
    off = (off + 255) & ~(size_t)255;
    void* p = base + off;
    off += bytes;
    return p;
  };

  __bf16* xb    = (__bf16*)alloc(4194304ull * 2);   // x in bf16
  __bf16* w1b   = (__bf16*)alloc(131072ull * 2);
  __bf16* g0awb = (__bf16*)alloc(131072ull * 2);
  __bf16* g0uwb = (__bf16*)alloc(131072ull * 2);
  __bf16* g1awb = (__bf16*)alloc(262144ull * 2);
  __bf16* g1uwb = (__bf16*)alloc(262144ull * 2);
  __bf16* cgwb  = (__bf16*)alloc(524288ull * 2);
  float*  y1    = (float*)alloc(2097152ull * 4);    // conv+BN output (2,256,4096)
  float*  mv2d  = (float*)alloc(512 * 4);
  float*  pos   = (float*)alloc(1048576ull * 4);    // masked-resampled support feats
  float*  colb  = (float*)alloc(4096 * 4);
  float*  s1    = (float*)alloc(512 * 4);
  float*  s2    = (float*)alloc(512 * 4);
  __bf16* xs_b  = (__bf16*)alloc(1048576ull * 2);
  __bf16* xq_b  = (__bf16*)alloc(1048576ull * 2);
  __bf16* xsn_b = (__bf16*)alloc(1048576ull * 2);
  __bf16* xqn_b = (__bf16*)alloc(1048576ull * 2);
  float*  bigf  = (float*)alloc(16777216ull * 4);   // 64MB scratch: A_s, A_q, R
  __bf16* AnS   = (__bf16*)alloc(16777216ull * 2);
  __bf16* AnQ   = (__bf16*)alloc(16777216ull * 2);
  __bf16* RTb   = (__bf16*)alloc(16777216ull * 2);
  __bf16* axT   = (__bf16*)alloc(2097152ull * 2);
  float*  ux    = (float*)alloc(2097152ull * 4);
  float*  emb1  = (float*)alloc(2097152ull * 4);
  float*  emb2  = (float*)alloc(2097152ull * 4);
  __bf16* e1n   = (__bf16*)alloc(2097152ull * 2);
  __bf16* e2n   = (__bf16*)alloc(2097152ull * 2);
  __bf16* e1T   = (__bf16*)alloc(2097152ull * 2);
  __bf16* c2b   = (__bf16*)alloc(4194304ull * 2);   // [emb2 | R^T emb1] bf16
  __bf16* e2b   = (__bf16*)alloc(2097152ull * 2);
  float*  part  = (float*)alloc(1024 * 4);
  float*  rmax  = (float*)alloc(256);

  auto EW = [](int n) { return dim3((unsigned)((n + 255) / 256)); };

  // ---- weight / input converts ----
  k_convert<<<EW(4194304), 256, 0, stream>>>(x, xb, 4194304);
  k_convert<<<EW(131072), 256, 0, stream>>>(conv1w, w1b, 131072);
  k_convert<<<EW(131072), 256, 0, stream>>>(g0aw, g0awb, 131072);
  k_convert<<<EW(131072), 256, 0, stream>>>(g0uw, g0uwb, 131072);
  k_convert<<<EW(262144), 256, 0, stream>>>(g1aw, g1awb, 262144);
  k_convert<<<EW(262144), 256, 0, stream>>>(g1uw, g1uwb, 262144);
  k_convert<<<EW(524288), 256, 0, stream>>>(cgw, cgwb, 524288);

  // ---- conv1 (1x1): y1[b] = W(256x512) @ x[b](512x4096) ----
  for (int b = 0; b < 2; ++b)
    k_gemm_bf16<<<dim3(32, 2), 256, 0, stream>>>(
        w1b, 512, xb + (size_t)b * 2097152, 4096, 0,
        y1 + (size_t)b * 1048576, 4096, nullptr, 0,
        nullptr, nullptr, 0, 512, EPI_F32);

  k_bn2d_reduce<<<256, 256, 0, stream>>>(y1, mv2d);
  k_bn2d_apply<<<EW(2097152), 256, 0, stream>>>(y1, mv2d, bn1g, bn1b);

  // ---- support branch mask resample (fused) ----
  k_posnode<<<EW(1048576), 256, 0, stream>>>(y1, mask, pos);

  // raw bf16 node features (flat reinterpret (4096,256))
  k_convert<<<EW(1048576), 256, 0, stream>>>(pos, xs_b, 1048576);
  k_convert<<<EW(1048576), 256, 0, stream>>>(y1 + 1048576, xq_b, 1048576);

  // ---- adjacency S: An_s (A.max cancels in column-normalization) ----
  hipMemsetAsync(colb, 0, 4096 * 4, stream);
  k_colsumsq<<<dim3(1, 8), 256, 0, stream>>>(pos, colb, 256, 512);
  k_colnorm_apply<<<EW(1048576), 256, 0, stream>>>(pos, colb, xsn_b, 1048576, 255);
  k_gemm_bf16<<<dim3(32, 32), 256, 0, stream>>>(xsn_b, 256, xsn_b, 256, 1,
      bigf, 4096, nullptr, 0, nullptr, nullptr, 0, 256, EPI_F32);
  hipMemsetAsync(colb, 0, 4096 * 4, stream);
  k_colabssum<<<dim3(16, 8), 256, 0, stream>>>(bigf, colb, 4096, 512);
  k_an_convert<<<EW(16777216), 256, 0, stream>>>(bigf, colb, AnS, 16777216);

  // ---- adjacency Q ----
  hipMemsetAsync(colb, 0, 4096 * 4, stream);
  k_colsumsq<<<dim3(1, 8), 256, 0, stream>>>(y1 + 1048576, colb, 256, 512);
  k_colnorm_apply<<<EW(1048576), 256, 0, stream>>>(y1 + 1048576, colb, xqn_b, 1048576, 255);
  k_gemm_bf16<<<dim3(32, 32), 256, 0, stream>>>(xqn_b, 256, xqn_b, 256, 1,
      bigf, 4096, nullptr, 0, nullptr, nullptr, 0, 256, EPI_F32);
  hipMemsetAsync(colb, 0, 4096 * 4, stream);
  k_colabssum<<<dim3(16, 8), 256, 0, stream>>>(bigf, colb, 4096, 512);
  k_an_convert<<<EW(16777216), 256, 0, stream>>>(bigf, colb, AnQ, 16777216);

  // ---- gnn layer 0: emb = An @ relu(x Wa^T + ba) + relu(x Wu^T + bu) ----
  // support
  k_gemm_bf16<<<dim3(4, 32), 256, 0, stream>>>(xs_b, 256, g0awb, 256, 1,
      nullptr, 0, axT, 4096, g0ab, nullptr, 0, 256, EPI_BIAS | EPI_RELU | EPI_B16 | EPI_B16T);
  k_gemm_bf16<<<dim3(4, 32), 256, 0, stream>>>(xs_b, 256, g0uwb, 256, 1,
      ux, 512, nullptr, 0, g0ub, nullptr, 0, 256, EPI_BIAS | EPI_RELU | EPI_F32);
  k_gemm_bf16<<<dim3(4, 32), 256, 0, stream>>>(AnS, 4096, axT, 4096, 1,
      emb1, 512, nullptr, 0, nullptr, ux, 512, 4096, EPI_ADD | EPI_F32);
  // query
  k_gemm_bf16<<<dim3(4, 32), 256, 0, stream>>>(xq_b, 256, g0awb, 256, 1,
      nullptr, 0, axT, 4096, g0ab, nullptr, 0, 256, EPI_BIAS | EPI_RELU | EPI_B16 | EPI_B16T);
  k_gemm_bf16<<<dim3(4, 32), 256, 0, stream>>>(xq_b, 256, g0uwb, 256, 1,
      ux, 512, nullptr, 0, g0ub, nullptr, 0, 256, EPI_BIAS | EPI_RELU | EPI_F32);
  k_gemm_bf16<<<dim3(4, 32), 256, 0, stream>>>(AnQ, 4096, axT, 4096, 1,
      emb2, 512, nullptr, 0, nullptr, ux, 512, 4096, EPI_ADD | EPI_F32);

  // ---- cross-graph R = normalize(emb1) @ normalize(emb2)^T, / R.max ----
  hipMemsetAsync(colb, 0, 512 * 4, stream);
  k_colsumsq<<<dim3(2, 8), 256, 0, stream>>>(emb1, colb, 512, 512);
  k_colnorm_apply<<<EW(2097152), 256, 0, stream>>>(emb1, colb, e1n, 2097152, 511);
  hipMemsetAsync(colb, 0, 512 * 4, stream);
  k_colsumsq<<<dim3(2, 8), 256, 0, stream>>>(emb2, colb, 512, 512);
  k_colnorm_apply<<<EW(2097152), 256, 0, stream>>>(emb2, colb, e2n, 2097152, 511);
  k_gemm_bf16<<<dim3(32, 32), 256, 0, stream>>>(e1n, 512, e2n, 512, 1,
      bigf, 4096, nullptr, 0, nullptr, nullptr, 0, 512, EPI_F32);
  k_max_partial<<<1024, 256, 0, stream>>>(bigf, part, 16777216);
  k_max_final<<<1, 256, 0, stream>>>(part, rmax, 1024);
  k_rT_convert<<<EW(16777216), 256, 0, stream>>>(bigf, rmax, RTb);

  // ---- cross fusion, emb2 path only (emb1_new is dead code) ----
  k_transpose_to_bf16<<<EW(2097152), 256, 0, stream>>>(emb1, e1T, 4096, 512);
  // c2 = [emb2 | R^T @ emb1] in bf16 (4096 x 1024)
  k_convert_strided<<<EW(2097152), 256, 0, stream>>>(emb2, 512, c2b, 1024, 4096, 512);
  k_gemm_bf16<<<dim3(4, 32), 256, 0, stream>>>(RTb, 4096, e1T, 4096, 1,
      nullptr, 0, c2b + 512, 1024, nullptr, nullptr, 0, 4096, EPI_B16);
  // Linear(1024->512) + bias
  k_gemm_bf16<<<dim3(4, 32), 256, 0, stream>>>(c2b, 1024, cgwb, 1024, 1,
      emb2, 512, nullptr, 0, cgb, nullptr, 0, 1024, EPI_BIAS | EPI_F32);
  // BN1d + ReLU (train stats over 4096 nodes)
  hipMemsetAsync(s1, 0, 512 * 4, stream);
  hipMemsetAsync(s2, 0, 512 * 4, stream);
  k_bn1d_reduce<<<dim3(2, 16), 256, 0, stream>>>(emb2, s1, s2);
  k_bn1d_apply<<<EW(2097152), 256, 0, stream>>>(emb2, s1, s2, cgbng, cgbnb);

  // ---- gnn layer 1, emb2 path only (final R & emb1 path are dead code) ----
  k_convert<<<EW(2097152), 256, 0, stream>>>(emb2, e2b, 2097152);
  k_gemm_bf16<<<dim3(4, 32), 256, 0, stream>>>(e2b, 512, g1awb, 512, 1,
      nullptr, 0, axT, 4096, g1ab, nullptr, 0, 512, EPI_BIAS | EPI_RELU | EPI_B16 | EPI_B16T);
  k_gemm_bf16<<<dim3(4, 32), 256, 0, stream>>>(e2b, 512, g1uwb, 512, 1,
      ux, 512, nullptr, 0, g1ub, nullptr, 0, 512, EPI_BIAS | EPI_RELU | EPI_F32);
  k_gemm_bf16<<<dim3(4, 32), 256, 0, stream>>>(AnQ, 4096, axT, 4096, 1,
      (float*)d_out, 512, nullptr, 0, nullptr, ux, 512, 4096, EPI_ADD | EPI_F32);
}
